// PatchEncoderLowMem_45578192945423
// MI455X (gfx1250) — matrix operand order repaired and checked by reading, not compiled, not run on hardware
//
#include <hip/hip_runtime.h>
#include <hip/hip_bf16.h>
#include <stdint.h>

// MalConv gated conv with S==K  ==> pure GEMM:
//   A (16384 x 4096) = embedded bytes, Bw (4096 x 128) = transposed w1/w2
//   g = (A@W1 + b1) * sigmoid(A@W2 + b2);  out[b,seg,c] = max over 32 rows.
// WG = 64 GEMM rows (2 segments) x 128 channels; wave = 16-channel tile.
// Double-buffered A slices; x staged via GLOBAL_LOAD_ASYNC_TO_LDS (ASYNCcnt),
// so B-fragment LOADcnt waits never stall on the HBM x stream.

typedef __attribute__((ext_vector_type(16))) _Float16 v16h;
typedef __attribute__((ext_vector_type(8)))  _Float16 v8h;
typedef __attribute__((ext_vector_type(8)))  float    v8f;
typedef __attribute__((ext_vector_type(4)))  int      v4i;
typedef __attribute__((ext_vector_type(4)))  unsigned int v4u;

#define T_LEN   1048576
#define K_W     512
#define R_TOT   4096          // K_W * 8 (GEMM reduction dim)
#define ROW_H   136           // 128 halfs + 8 pad (272 B row stride, 16B aligned)
#define WG_ROWS 64            // 4 M-tiles = 2 segments per workgroup
#define BUF_H   (WG_ROWS * ROW_H)
#define N_SLICE 32            // 4096 / 128; 16 bytes of x per row per slice

static __device__ __forceinline__ v16h cat8(v8h lo, v8h hi) {
  return __builtin_shufflevector(lo, hi, 0,1,2,3,4,5,6,7,8,9,10,11,12,13,14,15);
}

static __device__ __forceinline__ v8f wmma_f16(v16h a, v16h b, v8f c) {
  // D = A(16x32 f16) * B(32x16 f16) + C(16x16 f32)
  return __builtin_amdgcn_wmma_f32_16x16x32_f16(false, a, false, b, (short)0, c,
                                                false, false);
}

// Async global->LDS copy, 16B per lane, tracked by ASYNCcnt (not LOADcnt).
static __device__ __forceinline__ void async_load_b128(uint32_t lds_off,
                                                       uint32_t goff,
                                                       const void* gbase) {
  asm volatile("global_load_async_to_lds_b128 %0, %1, %2"
               :: "v"(lds_off), "v"(goff), "s"(gbase)
               : "memory");
}

static __device__ __forceinline__ void wait_async0() {
#if __has_builtin(__builtin_amdgcn_s_wait_asynccnt)
  __builtin_amdgcn_s_wait_asynccnt(0);
#else
  asm volatile("s_wait_asynccnt 0" ::: "memory");
#endif
}

// ---------------------------------------------------------------------------
// Pass 1: repack w1/w2 (C,E,K) f32 -> f16 fragments in WMMA B-register order.
// Fragment f = conv*1024 + rc*8 + ct  (rc: 32-wide K-chunk, ct: 16-col tile).
// Lane holds column n = lane&15 of tile ct, K values r = rc*32 + (lane>>4)*16 + i.
// ---------------------------------------------------------------------------
__global__ __launch_bounds__(256) void swizzle_w_kernel(
    const float* __restrict__ w1, const float* __restrict__ w2,
    _Float16* __restrict__ wsw) {
  const int t    = blockIdx.x * 256 + threadIdx.x;   // 0 .. 65535
  const int lane = t & 31;
  const int frag = t >> 5;                           // 0 .. 2047
  const int conv = frag >> 10;
  const int rem  = frag & 1023;
  const int rc   = rem >> 3;                         // 0 .. 127
  const int ct   = rem & 7;                          // 0 .. 7
  const int h    = lane >> 4;
  const int c    = ct * 16 + (lane & 15);
  const float* w = conv ? w2 : w1;                   // w[c*4096 + e*512 + k]
  alignas(16) _Float16 vals[16];
#pragma unroll
  for (int i = 0; i < 16; ++i) {
    const int r = rc * 32 + h * 16 + i;
    const int k = r >> 3;
    const int e = r & 7;
    vals[i] = (_Float16)w[c * R_TOT + e * K_W + k];
  }
  _Float16* dst = wsw + (size_t)frag * 512 + lane * 16;
  *(v8h*)dst       = *(const v8h*)&vals[0];
  *(v8h*)(dst + 8) = *(const v8h*)&vals[8];
}

// ---------------------------------------------------------------------------
// Pass 2: GEMM + gate + segment max. 256 WGs: one per (batch, segment-pair).
// ---------------------------------------------------------------------------
__global__ __launch_bounds__(256) void malconv_gemm_kernel(
    const int* __restrict__ x, const float* __restrict__ emb,
    const float* __restrict__ b1, const float* __restrict__ b2,
    const _Float16* __restrict__ wsw, float* __restrict__ out) {
  __shared__ alignas(16) _Float16 embl[257 * 8];      // f16 embedding table (4 KB)
  __shared__ alignas(16) _Float16 Asl[2 * BUF_H];     // 2 x (64 rows x 128 r)
  __shared__ alignas(16) int      xstg[2 * 1024];     // 2 x 4 KB raw x stage

  const int tid  = threadIdx.x;
  const int lane = tid & 31;
  const int wave = tid >> 5;        // column tile 0..7
  const int h    = lane >> 4;
  const int nn   = lane & 15;

  const int b  = blockIdx.x >> 5;   // batch
  const int sp = blockIdx.x & 31;   // segment pair: segments 2sp, 2sp+1

  // Embedding table -> LDS as f16 (8 KB f32 -> 4 KB f16)
  for (int v = tid; v < 257; v += 256) {
#pragma unroll
    for (int e = 0; e < 8; ++e)
      embl[v * 8 + e] = (_Float16)emb[v * 8 + e];
  }

  v8f acc1[4] = {{0}, {0}, {0}, {0}};   // conv1, M-tiles 0..3
  v8f acc2[4] = {{0}, {0}, {0}, {0}};   // conv2

  // Fill-phase role: thread handles (row, 4-byte group) of a 16-byte slice row
  const int row = tid >> 2;         // 0..63
  const int q   = tid & 3;          // bytes [q*4, q*4+4)
  // Global byte offset of this thread's x words for slice 0 (fits in 32 bits)
  const uint32_t gbyte0 =
      (uint32_t)(((size_t)b * T_LEN + (size_t)sp * WG_ROWS * K_W +
                  (size_t)row * K_W + q * 4) * 4);
  const int* xrow = x + (size_t)b * T_LEN + (size_t)sp * WG_ROWS * K_W
                      + (size_t)row * K_W;
  _Float16* scat_base = &Asl[row * ROW_H + q * 32];   // q*4 bytes * 8 halfs
  // LDS byte offset of this thread's 16B stage slot (aperture low 32 bits)
  const uint32_t stg_off = (uint32_t)(uintptr_t)(const void*)&xstg[tid * 4];

  // A-fragment per-lane offset (contiguous K layout: K runs [h*16,+16))
  const int aoff = nn * ROW_H + h * 16;

  __syncthreads();   // embl ready

  // ---- prologue: stage slice 0 directly; async-stage slice 1 ----
  {
    const v4i xv = __builtin_nontemporal_load((const v4i*)(xrow + q * 4));
    async_load_b128(stg_off + 4096, gbyte0 + 16, x);   // slice 1 -> xstg[1]
    v4u e0 = *(const v4u*)&embl[xv[0] * 8];
    v4u e1 = *(const v4u*)&embl[xv[1] * 8];
    v4u e2 = *(const v4u*)&embl[xv[2] * 8];
    v4u e3 = *(const v4u*)&embl[xv[3] * 8];
    v4u* dst = (v4u*)scat_base;
    dst[0] = e0; dst[1] = e1; dst[2] = e2; dst[3] = e3;
  }

  for (int kb = 0; kb < N_SLICE; ++kb) {
    __syncthreads();   // buf[kb&1] ready; previous compute finished

    // ---- compute slice kb from buf[kb&1]: 4 K-chunks of 32, 8 WMMAs each ----
    const _Float16* Ab = &Asl[(kb & 1) * BUF_H];
#pragma unroll
    for (int rcl = 0; rcl < 4; ++rcl) {
      const int rc = kb * 4 + rcl;
      const _Float16* pb1 = wsw + (size_t)(rc * 8 + wave) * 512 + lane * 16;
      const _Float16* pb2 = pb1 + (size_t)1024 * 512;      // conv 2 half of ws
      const v16h bf1 = cat8(*(const v8h*)pb1, *(const v8h*)(pb1 + 8));
      const v16h bf2 = cat8(*(const v8h*)pb2, *(const v8h*)(pb2 + 8));

#pragma unroll
      for (int mt = 0; mt < 4; ++mt) {
        const _Float16* pa = Ab + mt * (16 * ROW_H) + aoff + rcl * 32;
        const v16h af = cat8(*(const v8h*)pa, *(const v8h*)(pa + 8));
        acc1[mt] = wmma_f16(af, bf1, acc1[mt]);
        acc2[mt] = wmma_f16(af, bf2, acc2[mt]);
      }
    }

    // ---- scatter slice kb+1 (staged async a full iteration ago) ----
    if (kb + 1 < N_SLICE) {
      wait_async0();   // only ASYNCcnt; B-fragment LOADcnt waits untouched
      const v4i xv = *(const v4i*)&xstg[((kb + 1) & 1) * 1024 + tid * 4];
      if (kb + 2 < N_SLICE)   // async-stage slice kb+2 into the other buffer
        async_load_b128(stg_off + ((kb + 2) & 1) * 4096,
                        gbyte0 + (uint32_t)(kb + 2) * 16, x);
      v4u e0 = *(const v4u*)&embl[xv[0] * 8];
      v4u e1 = *(const v4u*)&embl[xv[1] * 8];
      v4u e2 = *(const v4u*)&embl[xv[2] * 8];
      v4u e3 = *(const v4u*)&embl[xv[3] * 8];
      v4u* dst = (v4u*)(scat_base + ((kb + 1) & 1) * BUF_H);
      dst[0] = e0; dst[1] = e1; dst[2] = e2; dst[3] = e3;
    }
  }

  // ---- epilogue: gate + max over 32 rows of each of the two segments ----
  const int c = wave * 16 + nn;
  const float bias1 = b1[c];
  const float bias2 = b2[c];
#pragma unroll
  for (int s = 0; s < 2; ++s) {
    float m = -3.402823466e38f;
#pragma unroll
    for (int mt = 2 * s; mt < 2 * s + 2; ++mt) {
#pragma unroll
      for (int j = 0; j < 8; ++j) {
        const float g = (acc1[mt][j] + bias1) *
                        (1.0f / (1.0f + __expf(-(acc2[mt][j] + bias2))));
        m = fmaxf(m, g);
      }
    }
    m = fmaxf(m, __shfl_xor(m, 16, 32));   // merge lane L with L+16 (same column)
    if (lane < 16)
      out[((size_t)b * 64 + sp * 2 + s) * 128 + c] = m;
  }
}

extern "C" void kernel_launch(void* const* d_in, const int* in_sizes, int n_in,
                              void* d_out, int out_size, void* d_ws, size_t ws_size,
                              hipStream_t stream) {
  (void)in_sizes; (void)n_in; (void)out_size; (void)ws_size;
  const int*   x   = (const int*)d_in[0];
  const float* emb = (const float*)d_in[1];
  const float* w1  = (const float*)d_in[2];
  const float* b1  = (const float*)d_in[3];
  const float* w2  = (const float*)d_in[4];
  const float* b2  = (const float*)d_in[5];
  float* out = (float*)d_out;
  _Float16* wsw = (_Float16*)d_ws;   // 2 MB: 2 convs x 4096 x 128 f16, fragment order

  swizzle_w_kernel<<<256, 256, 0, stream>>>(w1, w2, wsw);
  malconv_gemm_kernel<<<256, 256, 0, stream>>>(x, emb, b1, b2, wsw, out);
}
